// CroAttention_42236708389331
// MI455X (gfx1250) — matrix operand order repaired and checked by reading, compiled, not run on hardware
//
#include <hip/hip_runtime.h>

// ---------------------------------------------------------------------------
// CroAttention for MI455X (gfx1250, wave32, WMMA).
// Memory-bound op (~0.9 GB HBM traffic vs ~69 GFLOP) -> fuse everything,
// keep intermediates f16, v_wmma_f32_16x16x32_f16 for all matmuls, and use
// GLOBAL_LOAD_ASYNC_TO_LDS_B128 (ASYNCcnt) for f16 tile staging.
// Workspace requirement: ~98 MB.
// ---------------------------------------------------------------------------

#define SLICES 64   // b*t
#define CDIM   512
#define HEADS  8
#define DHEAD  64
#define LQ_    256
#define LK_    512

typedef __attribute__((ext_vector_type(16))) _Float16 v16h;
typedef __attribute__((ext_vector_type(8)))  _Float16 v8h;
typedef __attribute__((ext_vector_type(4)))  _Float16 v4h;
typedef __attribute__((ext_vector_type(8)))  float    v8f;
typedef __attribute__((ext_vector_type(4)))  int      v4i;

#if __has_builtin(__builtin_amdgcn_global_load_async_to_lds_b128)
#define ASYNC_LDS 1
typedef __attribute__((address_space(1))) v4i* as1_v4i_t;
typedef __attribute__((address_space(3))) v4i* as3_v4i_t;
#endif

// Async global->LDS 16-byte copy (no VGPR data round-trip, ASYNCcnt tracked).
// AS(1) ptr = flat 64-bit VA; AS(3) ptr = LDS offset = low 32 bits of the
// flat address (ISA 10.2: LDS aperture truncates to addr[31:0]).
static __device__ __forceinline__ void g2l_b128(const void* g, void* l) {
#ifdef ASYNC_LDS
  __builtin_amdgcn_global_load_async_to_lds_b128(
      (as1_v4i_t)(unsigned long long)g,
      (as3_v4i_t)(unsigned int)(unsigned long long)l, 0, 0);
#else
  *(v8h*)l = *(const v8h*)g;
#endif
}

static __device__ __forceinline__ void async_wait0() {
#ifdef ASYNC_LDS
#if __has_builtin(__builtin_amdgcn_s_wait_asynccnt)
  __builtin_amdgcn_s_wait_asynccnt(0);
#else
  asm volatile("s_wait_asynccnt 0x0" ::: "memory");
#endif
#endif
}

static __device__ __forceinline__ v8f wmma_f16(v16h a, v16h b, v8f c) {
  // D = A(16x32 f16) * B(32x16 f16) + C(16x16 f32)
  return __builtin_amdgcn_wmma_f32_16x16x32_f16(false, a, false, b,
                                                (short)0, c, false, false);
}

// A-matrix 16x32 f16 fragment (contiguous k order, same as load_b_frag):
//   row m = lane&15 ; elems e=0..15 -> k = (lane>>4)*16 + e
static __device__ __forceinline__ v16h load_a_frag(const _Float16* tile,
                                                   int ldk, int lane) {
  const _Float16* p = tile + (lane & 15) * ldk + ((lane >> 4) << 4);
  v8h lo = *(const v8h*)(p);
  v8h hi = *(const v8h*)(p + 8);
  v16h r;
#pragma unroll
  for (int i = 0; i < 8; ++i) { r[i] = lo[i]; r[i + 8] = hi[i]; }
  return r;
}

// B-matrix 32x16 f16 fragment, stored [n][k] with stride ldk:
//   col n = lane&15 ; k = (lane>>4)*16 + e (16 contiguous f16)
static __device__ __forceinline__ v16h load_b_frag(const _Float16* tile,
                                                   int ldk, int lane) {
  const _Float16* p = tile + (lane & 15) * ldk + ((lane >> 4) << 4);
  return *(const v16h*)(p);
}

// ---------------------------------------------------------------------------
__global__ void cvt_f32_f16(const float* __restrict__ s,
                            _Float16* __restrict__ d, int n) {
  int i = blockIdx.x * blockDim.x + threadIdx.x;
  if (i < n) d[i] = (_Float16)s[i];
}

// ---------------------------------------------------------------------------
// Fused projection GEMM: Y = X @ W^T + b, 64x64 output tile per WG (4 waves).
// MODE 0: X=f32 q,  epilogue: L2-norm over 64-wide head, *1/8, f16 head store
// MODE 1: X=f32 kv, epilogue: L2-norm, f16 store split into K-heads / V-heads
// MODE 2: X=f16 attn-out, epilogue: +bias +f32 residual, f32 store
// ---------------------------------------------------------------------------
template <int MODE>
__global__ __launch_bounds__(128) void proj_kernel(
    const float* __restrict__ X, const _Float16* __restrict__ A16,
    const _Float16* __restrict__ W16, const float* __restrict__ bias,
    const float* __restrict__ resid, _Float16* __restrict__ outH0,
    _Float16* __restrict__ outH1, float* __restrict__ outF, int R) {
  __shared__ __attribute__((aligned(32))) _Float16 As[64 * 64];
  __shared__ __attribute__((aligned(32))) _Float16 Bs[64 * 64];  // [n][k]

  const int rb = blockIdx.x, cb = blockIdx.y, slice = blockIdx.z;
  const int tid = threadIdx.x, w = tid >> 5, lane = tid & 31;
  const int grp = lane >> 4, nl = lane & 15;

  v8f acc[4] = {};

  for (int k0 = 0; k0 < CDIM; k0 += 64) {
    // ---- stage A tile (64 rows x 64 k) into LDS as f16 ----
    if (MODE < 2) {
      const float* Xb = X + ((size_t)slice * R + rb * 64) * CDIM + k0;
#pragma unroll
      for (int i = 0; i < 8; ++i) {
        int e = i * 128 + tid;              // float4 units
        int r = e >> 4, c = (e & 15) << 2;
        float4 f = *(const float4*)(Xb + (size_t)r * CDIM + c);
        v4h h;
        h[0] = (_Float16)f.x; h[1] = (_Float16)f.y;
        h[2] = (_Float16)f.z; h[3] = (_Float16)f.w;
        *(v4h*)(&As[r * 64 + c]) = h;
      }
    } else {
      const _Float16* Ab = A16 + ((size_t)slice * R + rb * 64) * CDIM + k0;
#pragma unroll
      for (int i = 0; i < 4; ++i) {
        int e = i * 128 + tid;              // v8h units
        int r = e >> 3, c = (e & 7) << 3;
        g2l_b128(Ab + (size_t)r * CDIM + c, &As[r * 64 + c]);
      }
    }
    // ---- stage B tile async: W16 row-major (Cout,K) -> Bs[n][k] ----
    {
      const _Float16* Wb = W16 + ((size_t)cb * 64) * CDIM + k0;
#pragma unroll
      for (int i = 0; i < 4; ++i) {
        int e = i * 128 + tid;
        int n = e >> 3, c = (e & 7) << 3;
        g2l_b128(Wb + (size_t)n * CDIM + c, &Bs[n * 64 + c]);
      }
    }
    async_wait0();
    __syncthreads();

#pragma unroll
    for (int kk = 0; kk < 64; kk += 32) {
      v16h a = load_a_frag(As + w * 16 * 64 + kk, 64, lane);
#pragma unroll
      for (int t = 0; t < 4; ++t) {
        v16h b = load_b_frag(Bs + t * 16 * 64 + kk, 64, lane);
        acc[t] = wmma_f16(a, b, acc[t]);
      }
    }
    __syncthreads();
  }

  // ---- epilogue ----
  float badd[4];
#pragma unroll
  for (int t = 0; t < 4; ++t) badd[t] = bias[cb * 64 + t * 16 + nl];

  if (MODE == 2) {
#pragma unroll
    for (int r = 0; r < 8; ++r) {
      int mg = rb * 64 + w * 16 + grp * 8 + r;
      size_t rowoff = ((size_t)slice * R + mg) * CDIM + cb * 64 + nl;
#pragma unroll
      for (int t = 0; t < 4; ++t)
        outF[rowoff + t * 16] = acc[t][r] + badd[t] + resid[rowoff + t * 16];
    }
  } else {
    const float scale = (MODE == 0) ? 0.125f : 1.0f;  // fold 1/sqrt(D) into Q
#pragma unroll
    for (int r = 0; r < 8; ++r) {
      float ss = 0.f;
#pragma unroll
      for (int t = 0; t < 4; ++t) {
        float v = acc[t][r] + badd[t];
        ss += v * v;
      }
      // row sum-of-squares lives on 16 lanes -> butterfly reduce (wave32)
      ss += __shfl_xor(ss, 1, 32);
      ss += __shfl_xor(ss, 2, 32);
      ss += __shfl_xor(ss, 4, 32);
      ss += __shfl_xor(ss, 8, 32);
      float inv = scale / fmaxf(sqrtf(ss), 1e-12f);
      int mg = rb * 64 + w * 16 + grp * 8 + r;
      _Float16* dst = outH0;
      int hidx = cb;
      if (MODE == 1 && hidx >= HEADS) { dst = outH1; hidx -= HEADS; }
      size_t off = (((size_t)slice * HEADS + hidx) * R + mg) * DHEAD + nl;
#pragma unroll
      for (int t = 0; t < 4; ++t)
        dst[off + t * 16] = (_Float16)((acc[t][r] + badd[t]) * inv);
    }
  }
}

// ---------------------------------------------------------------------------
// Flash-style cosine attention. One WG = (slice, head, 128 q-rows), 8 waves.
// K staged to LDS with async-to-LDS; V^T via transpose scatter; online
// softmax in registers; P relayout C->A through a wave-private LDS tile.
// ---------------------------------------------------------------------------
__global__ __launch_bounds__(256) void attn_kernel(
    const _Float16* __restrict__ qh, const _Float16* __restrict__ kh,
    const _Float16* __restrict__ vh, _Float16* __restrict__ xh) {
  __shared__ __attribute__((aligned(32))) _Float16 Ks[64 * 64];      // [key][d]
  __shared__ __attribute__((aligned(32))) _Float16 Vt[64 * 64];      // [d][key]
  __shared__ __attribute__((aligned(32))) _Float16 Ps[8 * 16 * 64];  // per-wave P

  const int rb = blockIdx.x, h = blockIdx.y, slice = blockIdx.z;
  const int tid = threadIdx.x, w = tid >> 5, lane = tid & 31;
  const int grp = lane >> 4, nl = lane & 15;

  // Q fragments for this wave's 16 rows, held for the whole K sweep
  const size_t qbase =
      (((size_t)slice * HEADS + h) * LQ_ + rb * 128 + w * 16) * DHEAD;
  v16h qf[2];
#pragma unroll
  for (int kc = 0; kc < 2; ++kc)
    qf[kc] = load_a_frag(qh + qbase + kc * 32, DHEAD, lane);

  v8f accO[4] = {};
  float mrun[8], lrun[8];
#pragma unroll
  for (int r = 0; r < 8; ++r) { mrun[r] = -1e30f; lrun[r] = 0.f; }

  const size_t kvbase = ((size_t)slice * HEADS + h) * (size_t)LK_ * DHEAD;

  for (int kb = 0; kb < LK_; kb += 64) {
    // ---- stage K block (async to LDS) and transposed V block ----
#pragma unroll
    for (int i = 0; i < 2; ++i) {
      int e = i * 256 + tid;                 // v8h units, 512 total
      int key = e >> 3, d0 = (e & 7) << 3;
      const size_t g = kvbase + (size_t)(kb + key) * DHEAD + d0;
      g2l_b128(kh + g, &Ks[key * 64 + d0]);
      v8h vv = *(const v8h*)(vh + g);
#pragma unroll
      for (int j = 0; j < 8; ++j) Vt[(d0 + j) * 64 + key] = vv[j];
    }
    // prefetch next K/V block into cache while this block computes
    if (kb + 64 < LK_) {
      const size_t pn = kvbase + (size_t)(kb + 64 + (tid >> 3)) * DHEAD +
                        ((tid & 7) << 3);
      __builtin_prefetch(kh + pn, 0, 1);
      __builtin_prefetch(vh + pn, 0, 1);
    }
    async_wait0();
    __syncthreads();

    // ---- S = Q K^T (scale already folded into Q) ----
    v8f s[4] = {};
#pragma unroll
    for (int kc = 0; kc < 2; ++kc) {
#pragma unroll
      for (int t = 0; t < 4; ++t) {
        v16h b = load_b_frag(Ks + t * 16 * 64 + kc * 32, 64, lane);
        s[t] = wmma_f16(qf[kc], b, s[t]);
      }
    }

    // ---- online softmax + materialize P in wave-private LDS ----
    _Float16* Pw = Ps + w * (16 * 64);
#pragma unroll
    for (int r = 0; r < 8; ++r) {
      float mb = fmaxf(fmaxf(s[0][r], s[1][r]), fmaxf(s[2][r], s[3][r]));
      mb = fmaxf(mb, __shfl_xor(mb, 1, 32));
      mb = fmaxf(mb, __shfl_xor(mb, 2, 32));
      mb = fmaxf(mb, __shfl_xor(mb, 4, 32));
      mb = fmaxf(mb, __shfl_xor(mb, 8, 32));
      float mnew = fmaxf(mrun[r], mb);
      float alpha = __expf(mrun[r] - mnew);
      mrun[r] = mnew;
      float ps = 0.f;
      int m = grp * 8 + r;
#pragma unroll
      for (int t = 0; t < 4; ++t) {
        float p = __expf(s[t][r] - mnew);
        ps += p;
        Pw[m * 64 + t * 16 + nl] = (_Float16)p;
      }
      ps += __shfl_xor(ps, 1, 32);
      ps += __shfl_xor(ps, 2, 32);
      ps += __shfl_xor(ps, 4, 32);
      ps += __shfl_xor(ps, 8, 32);
      lrun[r] = lrun[r] * alpha + ps;
#pragma unroll
      for (int t = 0; t < 4; ++t) accO[t][r] *= alpha;
    }

    // ---- O += P @ V  (same-wave LDS is in-order: RAW on Pw is safe) ----
#pragma unroll
    for (int kc = 0; kc < 2; ++kc) {
      v16h pa = load_a_frag(Pw + kc * 32, 64, lane);
#pragma unroll
      for (int t = 0; t < 4; ++t) {
        v16h b = load_b_frag(Vt + t * 16 * 64 + kc * 32, 64, lane);
        accO[t] = wmma_f16(pa, b, accO[t]);
      }
    }
    __syncthreads();
  }

  // ---- finalize: divide by row sum, store merged-head f16 ----
#pragma unroll
  for (int r = 0; r < 8; ++r) {
    float inv = 1.f / lrun[r];
    int mg = rb * 128 + w * 16 + grp * 8 + r;
    size_t off = ((size_t)slice * LQ_ + mg) * CDIM + h * DHEAD + nl;
#pragma unroll
    for (int t = 0; t < 4; ++t)
      xh[off + t * 16] = (_Float16)(accO[t][r] * inv);
  }
}

// ---------------------------------------------------------------------------
extern "C" void kernel_launch(void* const* d_in, const int* in_sizes, int n_in,
                              void* d_out, int out_size, void* d_ws,
                              size_t ws_size, hipStream_t stream) {
  (void)in_sizes; (void)n_in; (void)out_size; (void)ws_size;
  const float* q   = (const float*)d_in[0];
  const float* kv  = (const float*)d_in[1];
  const float* Wq  = (const float*)d_in[2];
  const float* bq  = (const float*)d_in[3];
  const float* Wkv = (const float*)d_in[4];
  const float* bkv = (const float*)d_in[5];
  const float* Wm  = (const float*)d_in[6];
  const float* bm  = (const float*)d_in[7];
  float* out = (float*)d_out;

  char* ws = (char*)d_ws;
  const size_t nWq = (size_t)CDIM * CDIM;        // 262144
  const size_t nWkv = (size_t)2 * CDIM * CDIM;   // 524288
  _Float16* Wq16  = (_Float16*)ws;
  _Float16* Wkv16 = (_Float16*)(ws + nWq * 2);
  _Float16* Wm16  = (_Float16*)(ws + (nWq + nWkv) * 2);
  size_t base = (nWq + nWkv + nWq) * 2;
  const size_t nQh = (size_t)SLICES * HEADS * LQ_ * DHEAD;  // 8M f16
  const size_t nKh = (size_t)SLICES * HEADS * LK_ * DHEAD;  // 16M f16
  _Float16* qh = (_Float16*)(ws + base);
  _Float16* kh = (_Float16*)(ws + base + nQh * 2);
  _Float16* vh = (_Float16*)(ws + base + nQh * 2 + nKh * 2);
  _Float16* xh = (_Float16*)(ws + base + nQh * 2 + 2 * nKh * 2);

  cvt_f32_f16<<<(int)((nWq + 255) / 256), 256, 0, stream>>>(Wq, Wq16, (int)nWq);
  cvt_f32_f16<<<(int)((nWkv + 255) / 256), 256, 0, stream>>>(Wkv, Wkv16,
                                                             (int)nWkv);
  cvt_f32_f16<<<(int)((nWq + 255) / 256), 256, 0, stream>>>(Wm, Wm16, (int)nWq);

  // Q projection + L2 norm + 1/sqrt(D): (256x512)@(512x512)^T per slice
  proj_kernel<0><<<dim3(LQ_ / 64, CDIM / 64, SLICES), 128, 0, stream>>>(
      q, nullptr, Wq16, bq, nullptr, qh, nullptr, nullptr, LQ_);
  // KV projection + L2 norm: (512x512)@(1024x512)^T per slice
  proj_kernel<1><<<dim3(LK_ / 64, 2 * CDIM / 64, SLICES), 128, 0, stream>>>(
      kv, nullptr, Wkv16, bkv, nullptr, kh, vh, nullptr, LK_);
  // Attention
  attn_kernel<<<dim3(LQ_ / 128, HEADS, SLICES), 256, 0, stream>>>(qh, kh, vh,
                                                                  xh);
  // Output projection + bias + residual
  proj_kernel<2><<<dim3(LQ_ / 64, CDIM / 64, SLICES), 128, 0, stream>>>(
      nullptr, xh, Wm16, bm, q, nullptr, nullptr, out, LQ_);
}
